// Discriminator_20426864460441
// MI455X (gfx1250) — matrix operand order repaired and checked by reading, not compiled, not run on hardware
//
#include <hip/hip_runtime.h>
#include <math.h>

typedef __attribute__((ext_vector_type(16))) _Float16 v16h;
typedef __attribute__((ext_vector_type(8)))  float    v8f;

#define HW   4096
#define CCH  8

// ---------------------------------------------------------------------------
// Generic 1x1 conv: in [B,CI,HW] -> out [B,CO,HW], optional bias
// ---------------------------------------------------------------------------
template<int CI, int CO>
__global__ void conv1x1_kernel(const float* __restrict__ in, const float* __restrict__ W,
                               const float* __restrict__ bias, float* __restrict__ out, int B)
{
    int idx = blockIdx.x * blockDim.x + threadIdx.x;
    if (idx >= B * HW) return;
    int b = idx / HW, pos = idx % HW;
    float xi[CI];
#pragma unroll
    for (int c = 0; c < CI; ++c) xi[c] = in[((size_t)b * CI + c) * HW + pos];
#pragma unroll
    for (int o = 0; o < CO; ++o) {
        float a = bias ? bias[o] : 0.f;
#pragma unroll
        for (int c = 0; c < CI; ++c) a += W[o * CI + c] * xi[c];
        out[((size_t)b * CO + o) * HW + pos] = a;
    }
}

// ---------------------------------------------------------------------------
// Per-channel sum / sumsq (deterministic tree reduction, one block per channel)
// ---------------------------------------------------------------------------
__global__ void bn_stats_kernel(const float* __restrict__ y, float* __restrict__ st,
                                int B, int C, int HWc)
{
    int c = blockIdx.x;
    __shared__ float s1[256], s2[256];
    float a = 0.f, q = 0.f;
    int tot = B * HWc;
    for (int i = threadIdx.x; i < tot; i += blockDim.x) {
        int bb = i / HWc, pos = i % HWc;
        float v = y[((size_t)bb * C + c) * HWc + pos];
        a += v; q += v * v;
    }
    s1[threadIdx.x] = a; s2[threadIdx.x] = q; __syncthreads();
    for (int s = 128; s > 0; s >>= 1) {
        if ((int)threadIdx.x < s) { s1[threadIdx.x] += s1[threadIdx.x + s]; s2[threadIdx.x] += s2[threadIdx.x + s]; }
        __syncthreads();
    }
    if (threadIdx.x == 0) { st[2 * c] = s1[0]; st[2 * c + 1] = s2[0]; }
}

// ---------------------------------------------------------------------------
// BN (batch stats) + ReLU
// ---------------------------------------------------------------------------
__global__ void bn_relu_kernel(const float* __restrict__ y, const float* __restrict__ st,
                               const float* __restrict__ g, const float* __restrict__ be,
                               float* __restrict__ out, int B, int C, int HWc)
{
    int idx = blockIdx.x * blockDim.x + threadIdx.x;
    int total = B * C * HWc;
    if (idx >= total) return;
    int c = (idx / HWc) % C;
    float n = (float)(B * HWc);
    float mean = st[2 * c] / n;
    float var  = st[2 * c + 1] / n - mean * mean;
    float inv  = rsqrtf(var + 1e-5f);
    float v = (y[idx] - mean) * inv * g[c] + be[c];
    out[idx] = fmaxf(v, 0.f);
}

// ---------------------------------------------------------------------------
// PAM QKV projection (1x1 conv 8->8 with bias), emit f16 in WMMA-friendly layouts
//   Qh,Kh: [B][pos][8]   Vh: [B][8][pos]
// ---------------------------------------------------------------------------
__global__ void pam_qkv_kernel(const float* __restrict__ pf,
                               const float* __restrict__ Wb, const float* __restrict__ bb,
                               const float* __restrict__ Wc, const float* __restrict__ bc,
                               const float* __restrict__ Wd, const float* __restrict__ bd,
                               _Float16* __restrict__ Qh, _Float16* __restrict__ Kh,
                               _Float16* __restrict__ Vh, int B)
{
    int idx = blockIdx.x * blockDim.x + threadIdx.x;
    if (idx >= B * HW) return;
    int b = idx / HW, pos = idx % HW;
    float xi[CCH];
#pragma unroll
    for (int c = 0; c < CCH; ++c) xi[c] = pf[((size_t)b * CCH + c) * HW + pos];
#pragma unroll
    for (int o = 0; o < CCH; ++o) {
        float aq = bb[o], ak = bc[o], av = bd[o];
#pragma unroll
        for (int c = 0; c < CCH; ++c) {
            aq += Wb[o * CCH + c] * xi[c];
            ak += Wc[o * CCH + c] * xi[c];
            av += Wd[o * CCH + c] * xi[c];
        }
        Qh[((size_t)b * HW + pos) * CCH + o] = (_Float16)aq;
        Kh[((size_t)b * HW + pos) * CCH + o] = (_Float16)ak;
        Vh[((size_t)b * CCH + o) * HW + pos] = (_Float16)av;
    }
}

// ---------------------------------------------------------------------------
// PAM flash attention: one wave per 16-row n-tile, WMMA f32_16x16x32_f16.
//   scores S[n,m] = sum_c Q[n][c] K[m][c]  (K padded 8->32 with zeros)
//   online softmax over m, O[n,c] = sum_m P[n,m] V[m][c]
//   out = alpha * (O/l) + pf
// ---------------------------------------------------------------------------
__global__ __launch_bounds__(256) void pam_attn_kernel(
    const _Float16* __restrict__ Qh, const _Float16* __restrict__ Kh,
    const _Float16* __restrict__ Vh, const float* __restrict__ pf,
    const float* __restrict__ alpha, float* __restrict__ out, int B)
{
    __shared__ _Float16 Pbuf[8][16 * 32];   // per-wave P tile (n x m_chunk)

    const int wave  = threadIdx.x >> 5;
    const int lane  = threadIdx.x & 31;
    const int lhalf = lane & 15;
    const bool hi   = lane >= 16;

    const int b    = blockIdx.x / 32;                 // 32 blocks per batch
    const int tile = (blockIdx.x % 32) * 8 + wave;    // 0..255
    const int n0   = tile * 16;

    // ---- A-fragment of Q (K=channel padded to 32; high half-wave all zero) ----
    v16h qa;
#pragma unroll
    for (int e = 0; e < 16; ++e) qa[e] = (_Float16)0;
    if (!hi) {
        const _Float16* qp = Qh + ((size_t)b * HW + (n0 + lhalf)) * CCH;
#pragma unroll
        for (int e = 0; e < 8; ++e) qa[e] = qp[e];
    }

    v8f acc, czero;
    float m_run[8], l_run[8];
#pragma unroll
    for (int r = 0; r < 8; ++r) { acc[r] = 0.f; czero[r] = 0.f; m_run[r] = -INFINITY; l_run[r] = 0.f; }

    for (int m0 = 0; m0 < HW; m0 += 32) {
        // ---- B-fragments of K for columns m0..m0+15 and m0+16..m0+31 ----
        v16h kb0, kb1;
#pragma unroll
        for (int e = 0; e < 16; ++e) { kb0[e] = (_Float16)0; kb1[e] = (_Float16)0; }
        if (!hi) {
            const _Float16* kp0 = Kh + ((size_t)b * HW + (m0 + lhalf)) * CCH;
            const _Float16* kp1 = Kh + ((size_t)b * HW + (m0 + 16 + lhalf)) * CCH;
#pragma unroll
            for (int e = 0; e < 8; ++e) { kb0[e] = kp0[e]; kb1[e] = kp1[e]; }
        }

        v8f S0 = __builtin_amdgcn_wmma_f32_16x16x32_f16(false, qa, false, kb0, (short)0, czero, false, false);
        v8f S1 = __builtin_amdgcn_wmma_f32_16x16x32_f16(false, qa, false, kb1, (short)0, czero, false, false);

        // ---- online softmax update (row = M, spread across 16-lane half) ----
        float P0[8], P1[8];
#pragma unroll
        for (int r = 0; r < 8; ++r) {
            float cm = fmaxf(S0[r], S1[r]);
#pragma unroll
            for (int off = 1; off <= 8; off <<= 1) cm = fmaxf(cm, __shfl_xor(cm, off, 32));
            float nm = fmaxf(m_run[r], cm);
            float sc = __expf(m_run[r] - nm);
            m_run[r] = nm;
            l_run[r] *= sc;
            acc[r]   *= sc;
            P0[r] = __expf(S0[r] - nm);
            P1[r] = __expf(S1[r] - nm);
            float rs = P0[r] + P1[r];
#pragma unroll
            for (int off = 1; off <= 8; off <<= 1) rs += __shfl_xor(rs, off, 32);
            l_run[r] += rs;
        }

        // ---- P (C-layout) -> LDS -> A-layout f16 ----
#pragma unroll
        for (int r = 0; r < 8; ++r) {
            int nrow = r + (hi ? 8 : 0);
            Pbuf[wave][nrow * 32 + lhalf]      = (_Float16)P0[r];
            Pbuf[wave][nrow * 32 + 16 + lhalf] = (_Float16)P1[r];
        }
        v16h pa;
#pragma unroll
        for (int e = 0; e < 8; ++e) {
            pa[e]     = Pbuf[wave][lhalf * 32 + (hi ? 8 + e  : e)];
            pa[8 + e] = Pbuf[wave][lhalf * 32 + (hi ? 24 + e : 16 + e)];
        }

        // ---- B-fragment of V: lane -> output channel, elements -> m ----
        v16h vb;
#pragma unroll
        for (int e = 0; e < 16; ++e) vb[e] = (_Float16)0;
        if (lhalf < CCH) {
            const _Float16* vp = Vh + ((size_t)b * CCH + lhalf) * HW + m0 + (hi ? 8 : 0);
#pragma unroll
            for (int e = 0; e < 16; ++e) vb[e] = vp[(e < 8) ? e : e + 8];
        }

        acc = __builtin_amdgcn_wmma_f32_16x16x32_f16(false, pa, false, vb, (short)0, acc, false, false);
    }

    // ---- write out = alpha * fe + pf ----
    float al = alpha[0];
    if (lhalf < CCH) {
        int c = lhalf;
#pragma unroll
        for (int r = 0; r < 8; ++r) {
            int n = n0 + r + (hi ? 8 : 0);
            size_t o = ((size_t)b * CCH + c) * HW + n;
            out[o] = al * (acc[r] / l_run[r]) + pf[o];
        }
    }
}

// ---------------------------------------------------------------------------
// Channel attention (one block per batch image)
// ---------------------------------------------------------------------------
__global__ __launch_bounds__(256) void cam_kernel(const float* __restrict__ cf,
                                                  const float* __restrict__ beta,
                                                  float* __restrict__ out, int B)
{
    int b = blockIdx.x;
    int wave = threadIdx.x >> 5, lane = threadIdx.x & 31;
    __shared__ float part[8][64];
    __shared__ float gram[64];
    __shared__ float attS[64];

    float accm[64];
#pragma unroll
    for (int p = 0; p < 64; ++p) accm[p] = 0.f;

    for (int pos = threadIdx.x; pos < HW; pos += 256) {
        float v[CCH];
#pragma unroll
        for (int c = 0; c < CCH; ++c) v[c] = cf[((size_t)b * CCH + c) * HW + pos];
#pragma unroll
        for (int c = 0; c < CCH; ++c)
#pragma unroll
            for (int d = 0; d < CCH; ++d) accm[c * CCH + d] += v[c] * v[d];
    }
#pragma unroll
    for (int p = 0; p < 64; ++p) {
        float s = accm[p];
#pragma unroll
        for (int off = 16; off >= 1; off >>= 1) s += __shfl_xor(s, off, 32);
        if (lane == 0) part[wave][p] = s;
    }
    __syncthreads();
    if (threadIdx.x < 64) {
        float s = 0.f;
#pragma unroll
        for (int w = 0; w < 8; ++w) s += part[w][threadIdx.x];
        gram[threadIdx.x] = s;
    }
    __syncthreads();
    if (threadIdx.x < CCH) {
        int c = threadIdx.x;
        float mx = -INFINITY, mn = INFINITY;
#pragma unroll
        for (int d = 0; d < CCH; ++d) { mx = fmaxf(mx, gram[c * CCH + d]); mn = fminf(mn, gram[c * CCH + d]); }
        float smax = mx - mn;        // max of (rowmax - score)
        float e[CCH]; float sum = 0.f;
#pragma unroll
        for (int d = 0; d < CCH; ++d) { e[d] = __expf((mx - gram[c * CCH + d]) - smax); sum += e[d]; }
#pragma unroll
        for (int d = 0; d < CCH; ++d) attS[c * CCH + d] = e[d] / sum;
    }
    __syncthreads();
    float bt = beta[0];
    for (int pos = threadIdx.x; pos < HW; pos += 256) {
        float v[CCH];
#pragma unroll
        for (int c = 0; c < CCH; ++c) v[c] = cf[((size_t)b * CCH + c) * HW + pos];
#pragma unroll
        for (int c = 0; c < CCH; ++c) {
            float fe = 0.f;
#pragma unroll
            for (int d = 0; d < CCH; ++d) fe += attS[c * CCH + d] * v[d];
            out[((size_t)b * CCH + c) * HW + pos] = bt * fe + v[c];
        }
    }
}

// ---------------------------------------------------------------------------
__global__ void add_kernel(const float* __restrict__ a, const float* __restrict__ bsrc,
                           float* __restrict__ out, int n)
{
    int i = blockIdx.x * blockDim.x + threadIdx.x;
    if (i < n) out[i] = a[i] + bsrc[i];
}

// ---------------------------------------------------------------------------
// 3x3 conv, pad 1, stride 1, with bias
// ---------------------------------------------------------------------------
template<int CI, int CO>
__global__ void conv3x3_kernel(const float* __restrict__ in, const float* __restrict__ W,
                               const float* __restrict__ bias, float* __restrict__ out,
                               int B, int H, int Wd)
{
    int idx = blockIdx.x * blockDim.x + threadIdx.x;
    int total = B * CO * H * Wd;
    if (idx >= total) return;
    int x  = idx % Wd;
    int y  = (idx / Wd) % H;
    int co = (idx / (Wd * H)) % CO;
    int b  = idx / (Wd * H * CO);
    float a = bias[co];
    for (int ci = 0; ci < CI; ++ci) {
        const float* ip = in + ((size_t)b * CI + ci) * H * Wd;
        const float* wp = W + ((size_t)co * CI + ci) * 9;
#pragma unroll
        for (int ky = 0; ky < 3; ++ky) {
            int iy = y + ky - 1;
            if (iy < 0 || iy >= H) continue;
#pragma unroll
            for (int kx = 0; kx < 3; ++kx) {
                int ix = x + kx - 1;
                if (ix < 0 || ix >= Wd) continue;
                a += ip[iy * Wd + ix] * wp[ky * 3 + kx];
            }
        }
    }
    out[idx] = a;
}

// ---------------------------------------------------------------------------
// BN + ReLU + non-overlapping maxpool(k)
// ---------------------------------------------------------------------------
__global__ void bn_relu_pool_kernel(const float* __restrict__ y, const float* __restrict__ st,
                                    const float* __restrict__ g, const float* __restrict__ be,
                                    float* __restrict__ out, int B, int C, int H, int W, int k)
{
    int Ho = H / k, Wo = W / k;
    int total = B * C * Ho * Wo;
    int idx = blockIdx.x * blockDim.x + threadIdx.x;
    if (idx >= total) return;
    int xo = idx % Wo;
    int yo = (idx / Wo) % Ho;
    int c  = (idx / (Wo * Ho)) % C;
    int b  = idx / (Wo * Ho * C);
    float n = (float)(B * H * W);
    float mean = st[2 * c] / n;
    float var  = st[2 * c + 1] / n - mean * mean;
    float inv  = rsqrtf(var + 1e-5f);
    float gc = g[c], bc = be[c];
    float mx = -INFINITY;
    for (int ky = 0; ky < k; ++ky)
        for (int kx = 0; kx < k; ++kx) {
            float v = y[(((size_t)b * C + c) * H + yo * k + ky) * W + xo * k + kx];
            v = fmaxf((v - mean) * inv * gc + bc, 0.f);
            mx = fmaxf(mx, v);
        }
    out[idx] = mx;
}

// ReLU + maxpool(k) (no BN) for the last conv stage
__global__ void relu_pool_kernel(const float* __restrict__ y, float* __restrict__ out,
                                 int B, int C, int H, int W, int k)
{
    int Ho = H / k, Wo = W / k;
    int total = B * C * Ho * Wo;
    int idx = blockIdx.x * blockDim.x + threadIdx.x;
    if (idx >= total) return;
    int xo = idx % Wo;
    int yo = (idx / Wo) % Ho;
    int c  = (idx / (Wo * Ho)) % C;
    int b  = idx / (Wo * Ho * C);
    float mx = -INFINITY;
    for (int ky = 0; ky < k; ++ky)
        for (int kx = 0; kx < k; ++kx) {
            float v = y[(((size_t)b * C + c) * H + yo * k + ky) * W + xo * k + kx];
            mx = fmaxf(mx, fmaxf(v, 0.f));
        }
    out[idx] = mx;
}

// ---------------------------------------------------------------------------
__global__ void fc1_kernel(const float* __restrict__ feat, const float* __restrict__ W,
                           const float* __restrict__ bias, float* __restrict__ out)
{
    int b = blockIdx.x, j = threadIdx.x;   // 64 threads
    float a = bias[j];
    for (int i = 0; i < 512; ++i) a += W[j * 512 + i] * feat[b * 512 + i];
    out[b * 64 + j] = fmaxf(a, 0.f);
}

__global__ void fc2_kernel(const float* __restrict__ h, const float* __restrict__ W,
                           const float* __restrict__ bias, float* __restrict__ out)
{
    int b = blockIdx.x, j = threadIdx.x;
    if (j < 6) {
        float a = bias[j];
        for (int i = 0; i < 64; ++i) a += W[j * 64 + i] * h[b * 64 + i];
        out[b * 6 + j] = a;
    }
}

// ---------------------------------------------------------------------------
extern "C" void kernel_launch(void* const* d_in, const int* in_sizes, int n_in,
                              void* d_out, int out_size, void* d_ws, size_t ws_size,
                              hipStream_t stream)
{
    const float* X    = (const float*)d_in[0];
    const float* Wp1  = (const float*)d_in[1];
    const float* gp1  = (const float*)d_in[2];
    const float* bp1  = (const float*)d_in[3];
    const float* pWb  = (const float*)d_in[4];
    const float* pbb  = (const float*)d_in[5];
    const float* pWc  = (const float*)d_in[6];
    const float* pbc  = (const float*)d_in[7];
    const float* pWd  = (const float*)d_in[8];
    const float* pbd  = (const float*)d_in[9];
    const float* alpha= (const float*)d_in[10];
    const float* beta = (const float*)d_in[11];
    const float* Wp2  = (const float*)d_in[12];
    const float* gp2  = (const float*)d_in[13];
    const float* bp2  = (const float*)d_in[14];
    const float* Wc1  = (const float*)d_in[15];
    const float* gc1  = (const float*)d_in[16];
    const float* bc1  = (const float*)d_in[17];
    const float* Wc2  = (const float*)d_in[18];
    const float* gc2  = (const float*)d_in[19];
    const float* bc2  = (const float*)d_in[20];
    const float* W1   = (const float*)d_in[21];
    const float* b1   = (const float*)d_in[22];
    const float* g1   = (const float*)d_in[23];
    const float* be1  = (const float*)d_in[24];
    const float* W2   = (const float*)d_in[25];
    const float* b2   = (const float*)d_in[26];
    const float* g2   = (const float*)d_in[27];
    const float* be2  = (const float*)d_in[28];
    const float* W3   = (const float*)d_in[29];
    const float* b3   = (const float*)d_in[30];
    const float* g3   = (const float*)d_in[31];
    const float* be3  = (const float*)d_in[32];
    const float* W4   = (const float*)d_in[33];
    const float* b4   = (const float*)d_in[34];
    const float* Wfc1 = (const float*)d_in[35];
    const float* bfc1 = (const float*)d_in[36];
    const float* Wfc2 = (const float*)d_in[37];
    const float* bfc2 = (const float*)d_in[38];
    float* outp = (float*)d_out;

    const int B = in_sizes[0] / (2 * HW);   // 16
    const int NPOS = B * HW;                // 65536

    // ---- workspace bump allocation (floats) ----
    float* ws      = (float*)d_ws;
    float* bufA    = ws;                    // 1,048,576 f (max: [B,16,4096])
    float* bufB    = bufA + 1048576;        // 1,048,576 f
    float* pfb     = bufB + 1048576;        // 524,288 f  [B,8,4096]
    float* cfb     = pfb  + 524288;
    float* pam_out = cfb  + 524288;
    float* cam_out = pam_out + 524288;
    float* st      = cam_out + 524288;      // 256 f (per-channel sum/sumsq)
    float* fcb     = st + 256;              // B*64
    _Float16* Qh   = (_Float16*)(fcb + 1024);
    _Float16* Kh   = Qh + (size_t)B * HW * CCH;
    _Float16* Vh   = Kh + (size_t)B * HW * CCH;

    const int TPB = 256;
    auto blk = [](int n) { return (n + 255) / 256; };

    // ---- stage 1: 1x1 convs (no bias) + BN + ReLU for both branches ----
    conv1x1_kernel<2, 8><<<blk(NPOS), TPB, 0, stream>>>(X, Wp1, nullptr, bufA, B);
    bn_stats_kernel<<<8, TPB, 0, stream>>>(bufA, st, B, 8, HW);
    bn_relu_kernel<<<blk(NPOS * 8), TPB, 0, stream>>>(bufA, st, gp1, bp1, pfb, B, 8, HW);

    conv1x1_kernel<2, 8><<<blk(NPOS), TPB, 0, stream>>>(X, Wc1, nullptr, bufB, B);
    bn_stats_kernel<<<8, TPB, 0, stream>>>(bufB, st, B, 8, HW);
    bn_relu_kernel<<<blk(NPOS * 8), TPB, 0, stream>>>(bufB, st, gc1, bc1, cfb, B, 8, HW);

    // ---- PAM: QKV projection + flash attention (WMMA) ----
    pam_qkv_kernel<<<blk(NPOS), TPB, 0, stream>>>(pfb, pWb, pbb, pWc, pbc, pWd, pbd, Qh, Kh, Vh, B);
    pam_attn_kernel<<<B * 32, TPB, 0, stream>>>(Qh, Kh, Vh, pfb, alpha, pam_out, B);

    // ---- CAM ----
    cam_kernel<<<B, TPB, 0, stream>>>(cfb, beta, cam_out, B);

    // ---- stage 2: 1x1 convs + BN + ReLU ----
    conv1x1_kernel<8, 8><<<blk(NPOS), TPB, 0, stream>>>(pam_out, Wp2, nullptr, bufA, B);
    bn_stats_kernel<<<8, TPB, 0, stream>>>(bufA, st, B, 8, HW);
    bn_relu_kernel<<<blk(NPOS * 8), TPB, 0, stream>>>(bufA, st, gp2, bp2, pfb, B, 8, HW);

    conv1x1_kernel<8, 8><<<blk(NPOS), TPB, 0, stream>>>(cam_out, Wc2, nullptr, bufB, B);
    bn_stats_kernel<<<8, TPB, 0, stream>>>(bufB, st, B, 8, HW);
    bn_relu_kernel<<<blk(NPOS * 8), TPB, 0, stream>>>(bufB, st, gc2, bc2, cfb, B, 8, HW);

    // ---- fusion ----
    add_kernel<<<blk(NPOS * 8), TPB, 0, stream>>>(pfb, cfb, pam_out, NPOS * 8);

    // ---- L1: 3x3 conv 8->16 (64x64), BN, ReLU, pool2 -> [B,16,32,32] ----
    conv3x3_kernel<8, 16><<<blk(B * 16 * 4096), TPB, 0, stream>>>(pam_out, W1, b1, bufA, B, 64, 64);
    bn_stats_kernel<<<16, TPB, 0, stream>>>(bufA, st, B, 16, 4096);
    bn_relu_pool_kernel<<<blk(B * 16 * 1024), TPB, 0, stream>>>(bufA, st, g1, be1, bufB, B, 16, 64, 64, 2);

    // ---- L2: 16->32 (32x32), BN, ReLU, pool4 -> [B,32,8,8] ----
    conv3x3_kernel<16, 32><<<blk(B * 32 * 1024), TPB, 0, stream>>>(bufB, W2, b2, bufA, B, 32, 32);
    bn_stats_kernel<<<32, TPB, 0, stream>>>(bufA, st, B, 32, 1024);
    bn_relu_pool_kernel<<<blk(B * 32 * 64), TPB, 0, stream>>>(bufA, st, g2, be2, bufB, B, 32, 32, 32, 4);

    // ---- L3: 32->64 (8x8), BN, ReLU, pool2 -> [B,64,4,4] ----
    conv3x3_kernel<32, 64><<<blk(B * 64 * 64), TPB, 0, stream>>>(bufB, W3, b3, bufA, B, 8, 8);
    bn_stats_kernel<<<64, TPB, 0, stream>>>(bufA, st, B, 64, 64);
    bn_relu_pool_kernel<<<blk(B * 64 * 16), TPB, 0, stream>>>(bufA, st, g3, be3, bufB, B, 64, 8, 8, 2);

    // ---- L4: 64->128 (4x4), ReLU, pool2 -> [B,128,2,2] = [B,512] ----
    conv3x3_kernel<64, 128><<<blk(B * 128 * 16), TPB, 0, stream>>>(bufB, W4, b4, bufA, B, 4, 4);
    relu_pool_kernel<<<blk(B * 128 * 4), TPB, 0, stream>>>(bufA, bufB, B, 128, 4, 4, 2);

    // ---- FC head ----
    fc1_kernel<<<B, 64, 0, stream>>>(bufB, Wfc1, bfc1, fcb);
    fc2_kernel<<<B, 32, 0, stream>>>(fcb, Wfc2, bfc2, outp);
}
